// LSTMModelAttention_79053168050582
// MI455X (gfx1250) — matrix operand order repaired and checked by reading, not compiled, not run on hardware
//
#include <hip/hip_runtime.h>
#include <hip/hip_bf16.h>

#define BATCH 4096
#define SEQ_T 365
#define FEAT  32
#define HID   256
#define ROWS  16    // batch rows per wave (one wave per workgroup)
#define LDH   264   // padded LDS row stride for HID-wide arrays (bank-conflict-free b128)
#define LDF   40    // padded LDS row stride for FEAT-wide arrays

typedef __attribute__((ext_vector_type(16))) __bf16       bf16x16;
typedef __attribute__((ext_vector_type(8)))  float        f32x8;
typedef __attribute__((ext_vector_type(4)))  unsigned int u32x4;

union BV { bf16x16 v; u32x4 q[2]; };

__device__ __forceinline__ unsigned short f2bf(float f) {
  unsigned int u = __float_as_uint(f);
  u += 0x7FFFu + ((u >> 16) & 1u);          // round-to-nearest-even
  return (unsigned short)(u >> 16);
}
__device__ __forceinline__ float bf2f(unsigned short s) {
  return __uint_as_float(((unsigned int)s) << 16);
}
__device__ __forceinline__ float sigmoidf_(float x) {
  return 1.0f / (1.0f + __expf(-x));
}

// A tile (16x32 bf16, rows = batch) from LDS per ISA 16-bit A layout.
__device__ __forceinline__ bf16x16 load_a(const unsigned short* base, int stride,
                                          int kbase, int lane) {
  const int row = lane & 15;
  const int hi  = lane >> 4;
  BV r;
  const unsigned short* p = base + row * stride + kbase + hi * 8;
  r.q[0] = *(const u32x4*)(p);        // K = kbase + hi*8 .. +7
  r.q[1] = *(const u32x4*)(p + 16);   // K = kbase + 16 + hi*8 .. +7
  return r.v;
}

// B tile (32x16 bf16) from [N][K] row-major bf16 weights in global memory.
__device__ __forceinline__ bf16x16 load_b(const unsigned short* w, int ldk,
                                          int nbase, int kbase, int lane) {
  const int col  = lane & 15;
  const int koff = (lane >> 4) * 8;
  BV r;
  const u32x4* p = (const u32x4*)(w + (size_t)(nbase + col) * ldk + kbase + koff);
  r.q[0] = p[0];
  r.q[1] = p[2];
  return r.v;
}

__device__ __forceinline__ f32x8 wmma_bf16(bf16x16 a, bf16x16 b, f32x8 c) {
  return __builtin_amdgcn_wmma_f32_16x16x32_bf16(false, a, false, b,
                                                 (short)0, c, false, false);
}

__global__ void cvt_f32_bf16_kernel(const float* __restrict__ src,
                                    unsigned short* __restrict__ dst, int n) {
  int i = blockIdx.x * blockDim.x + threadIdx.x;
  if (i < n) dst[i] = f2bf(src[i]);
}

// One wave (32 threads) per block; each wave owns ROWS=16 batch rows for all T steps.
__global__ __launch_bounds__(32)
void lstm_attn_persistent_kernel(
    const float* __restrict__ x,        // [B, T, F]
    const float* __restrict__ W_b,      // [H]
    const float* __restrict__ U_b,      // [H]
    const float* __restrict__ V_b,      // [F]
    const float* __restrict__ bih,      // [4H]
    const float* __restrict__ bhh,      // [4H]
    const float* __restrict__ fc_w,     // [H]
    const float* __restrict__ fc_b,     // [1]
    const unsigned short* __restrict__ Wbf,   // [H][2H]   bf16
    const unsigned short* __restrict__ Ubf,   // [H][F]    bf16
    const unsigned short* __restrict__ Vbf,   // [F][H]    bf16
    const unsigned short* __restrict__ Wihbf, // [4H][F]   bf16
    const unsigned short* __restrict__ Whhbf, // [4H][H]   bf16
    float* __restrict__ out)            // [B] pred ++ [T*B*F] attn
{
  __shared__ unsigned short h_buf[2][ROWS * LDH]; // bf16 h, double-buffered
  __shared__ unsigned short c_bf[ROWS * LDH];     // bf16 c (GEMM operand)
  __shared__ float          c_f32[ROWS * LDH];    // f32 c (recurrence master)
  __shared__ unsigned short sact[ROWS * LDH];     // tanh(...) bf16
  __shared__ unsigned short xw[ROWS * LDF];       // x_t tile bf16
  __shared__ unsigned short wxb[ROWS * LDF];      // attn*x_t bf16
  __shared__ float          sc[ROWS * LDF];       // scores f32

  const int lane    = threadIdx.x;
  const int rowbase = blockIdx.x * ROWS;

  for (int i = lane; i < ROWS * LDH; i += 32) {
    h_buf[0][i] = 0; h_buf[1][i] = 0; c_bf[i] = 0; c_f32[i] = 0.0f;
  }
  __syncthreads();

  #pragma unroll 1
  for (int t = 0; t < SEQ_T; ++t) {
    const unsigned short* hcur = h_buf[t & 1];
    unsigned short*       hnew = h_buf[(t + 1) & 1];

    // ---- stage x_t tile as bf16 in LDS ----
    {
      const int row = lane & 15, half = lane >> 4;
      const float* xp = x + ((size_t)(rowbase + row) * SEQ_T + t) * FEAT + half * 16;
      #pragma unroll
      for (int j = 0; j < 16; ++j) xw[row * LDF + half * 16 + j] = f2bf(xp[j]);
    }
    __syncthreads();

    // ---- GEMM1: S1[16,256] = [h|c]@W_wT + x_t@U_wT + (W_b+U_b) ----
    // Two N-passes of 128 columns (8 accumulator tiles each) to bound VGPR pressure.
    #pragma unroll 1
    for (int p = 0; p < 2; ++p) {
      const int n0 = p * 128;                // column base of this pass
      f32x8 acc[8];
      #pragma unroll
      for (int n = 0; n < 8; ++n) {
        const int col = n0 + n * 16 + (lane & 15);
        const float b = W_b[col] + U_b[col];
        #pragma unroll
        for (int r = 0; r < 8; ++r) acc[n][r] = b;
      }
      #pragma unroll 1
      for (int kt = 0; kt < 8; ++kt) {       // K = 0..255 : h
        const bf16x16 A = load_a(hcur, LDH, kt * 32, lane);
        #pragma unroll
        for (int n = 0; n < 8; ++n)
          acc[n] = wmma_bf16(A, load_b(Wbf, 2 * HID, n0 + n * 16, kt * 32, lane), acc[n]);
      }
      #pragma unroll 1
      for (int kt = 0; kt < 8; ++kt) {       // K = 256..511 : c
        const bf16x16 A = load_a(c_bf, LDH, kt * 32, lane);
        #pragma unroll
        for (int n = 0; n < 8; ++n)
          acc[n] = wmma_bf16(A, load_b(Wbf, 2 * HID, n0 + n * 16, 256 + kt * 32, lane), acc[n]);
      }
      {                                      // + x_t @ U_wT (K=32)
        const bf16x16 A = load_a(xw, LDF, 0, lane);
        #pragma unroll
        for (int n = 0; n < 8; ++n)
          acc[n] = wmma_bf16(A, load_b(Ubf, FEAT, n0 + n * 16, 0, lane), acc[n]);
      }
      // tanh -> bf16 LDS (C layout: m = (lane>>4)*8 + r, col = n0 + n*16 + (lane&15))
      #pragma unroll
      for (int n = 0; n < 8; ++n) {
        #pragma unroll
        for (int r = 0; r < 8; ++r) {
          const int m = (lane >> 4) * 8 + r;
          sact[m * LDH + n0 + n * 16 + (lane & 15)] = f2bf(tanhf(acc[n][r]));
        }
      }
    }
    __syncthreads();

    // ---- GEMM2: scores[16,32] = tanh(S1) @ V_wT + V_b ----
    {
      f32x8 s0, s1;
      const float b0 = V_b[lane & 15], b1 = V_b[16 + (lane & 15)];
      #pragma unroll
      for (int r = 0; r < 8; ++r) { s0[r] = b0; s1[r] = b1; }
      #pragma unroll 1
      for (int kt = 0; kt < 8; ++kt) {
        const bf16x16 A = load_a(sact, LDH, kt * 32, lane);
        s0 = wmma_bf16(A, load_b(Vbf, HID, 0,  kt * 32, lane), s0);
        s1 = wmma_bf16(A, load_b(Vbf, HID, 16, kt * 32, lane), s1);
      }
      #pragma unroll
      for (int r = 0; r < 8; ++r) {
        const int m = (lane >> 4) * 8 + r, n = lane & 15;
        sc[m * LDF + n]      = s0[r];
        sc[m * LDF + 16 + n] = s1[r];
      }
    }
    __syncthreads();

    // ---- softmax over F=32 (lanes 0..15), attn -> global (NT), wx = attn*x_t ----
    if (lane < 16) {
      const int m = lane;
      float mx = -INFINITY;
      #pragma unroll
      for (int j = 0; j < 32; ++j) mx = fmaxf(mx, sc[m * LDF + j]);
      float sum = 0.0f;
      #pragma unroll
      for (int j = 0; j < 32; ++j) sum += __expf(sc[m * LDF + j] - mx);
      const float inv = 1.0f / sum;
      const float* xp = x + ((size_t)(rowbase + m) * SEQ_T + t) * FEAT;
      float* ao = out + BATCH + ((size_t)t * BATCH + rowbase + m) * FEAT;
      #pragma unroll
      for (int j = 0; j < 32; ++j) {
        const float a = __expf(sc[m * LDF + j] - mx) * inv;
        __builtin_nontemporal_store(a, ao + j);     // streaming, never re-read
        wxb[m * LDF + j] = f2bf(a * xp[j]);
      }
    }
    __syncthreads();

    // ---- GEMM3 + fused LSTM: 8 column chunks of 32, all 4 gates live per chunk ----
    #pragma unroll 1
    for (int ch = 0; ch < 8; ++ch) {
      const int j0 = ch * 32;
      f32x8 g[4][2];
      #pragma unroll
      for (int gi = 0; gi < 4; ++gi)
        #pragma unroll
        for (int nt = 0; nt < 2; ++nt) {
          const int idx = gi * HID + j0 + nt * 16 + (lane & 15);
          const float b = bih[idx] + bhh[idx];
          #pragma unroll
          for (int r = 0; r < 8; ++r) g[gi][nt][r] = b;
        }
      {                                      // wx @ WihT (K=32)
        const bf16x16 A = load_a(wxb, LDF, 0, lane);
        #pragma unroll
        for (int gi = 0; gi < 4; ++gi)
          #pragma unroll
          for (int nt = 0; nt < 2; ++nt)
            g[gi][nt] = wmma_bf16(A,
              load_b(Wihbf, FEAT, gi * HID + j0 + nt * 16, 0, lane), g[gi][nt]);
      }
      #pragma unroll 1
      for (int kt = 0; kt < 8; ++kt) {       // h @ WhhT (K=256)
        const bf16x16 A = load_a(hcur, LDH, kt * 32, lane);
        #pragma unroll
        for (int gi = 0; gi < 4; ++gi)
          #pragma unroll
          for (int nt = 0; nt < 2; ++nt)
            g[gi][nt] = wmma_bf16(A,
              load_b(Whhbf, HID, gi * HID + j0 + nt * 16, kt * 32, lane), g[gi][nt]);
      }
      // fused elementwise LSTM update for columns j0 .. j0+31
      #pragma unroll
      for (int nt = 0; nt < 2; ++nt) {
        #pragma unroll
        for (int r = 0; r < 8; ++r) {
          const int m   = (lane >> 4) * 8 + r;
          const int col = j0 + nt * 16 + (lane & 15);
          const float iv = sigmoidf_(g[0][nt][r]);
          const float fv = sigmoidf_(g[1][nt][r]);
          const float gv = tanhf(g[2][nt][r]);
          const float ov = sigmoidf_(g[3][nt][r]);
          const float cn = fv * c_f32[m * LDH + col] + iv * gv;
          const float hn = ov * tanhf(cn);
          c_f32[m * LDH + col] = cn;
          c_bf [m * LDH + col] = f2bf(cn);
          hnew [m * LDH + col] = f2bf(hn);
        }
      }
    }
    __syncthreads();
  }

  // ---- pred = h_T @ fc_wT + fc_b ----
  {
    const unsigned short* hfin = h_buf[SEQ_T & 1];
    const int row = lane & 15, half = lane >> 4;
    float s = 0.0f;
    #pragma unroll 1
    for (int j = 0; j < 128; ++j)
      s += bf2f(hfin[row * LDH + half * 128 + j]) * fc_w[half * 128 + j];
    s += __shfl_xor(s, 16, 32);
    if (half == 0) out[rowbase + row] = s + fc_b[0];
  }
}

extern "C" void kernel_launch(void* const* d_in, const int* in_sizes, int n_in,
                              void* d_out, int out_size, void* d_ws, size_t ws_size,
                              hipStream_t stream) {
  const float* x    = (const float*)d_in[0];
  const float* W_w  = (const float*)d_in[1];
  const float* W_b  = (const float*)d_in[2];
  const float* U_w  = (const float*)d_in[3];
  const float* U_b  = (const float*)d_in[4];
  const float* V_w  = (const float*)d_in[5];
  const float* V_b  = (const float*)d_in[6];
  const float* Wih  = (const float*)d_in[7];
  const float* Whh  = (const float*)d_in[8];
  const float* bih  = (const float*)d_in[9];
  const float* bhh  = (const float*)d_in[10];
  const float* fc_w = (const float*)d_in[11];
  const float* fc_b = (const float*)d_in[12];
  float* out = (float*)d_out;

  // bf16 weight copies in workspace
  unsigned short* ws    = (unsigned short*)d_ws;
  unsigned short* Wbf   = ws;                         // 256*512
  unsigned short* Ubf   = Wbf   + 256 * 512;          // 256*32
  unsigned short* Vbf   = Ubf   + 256 * 32;           // 32*256
  unsigned short* Wihbf = Vbf   + 32 * 256;           // 1024*32
  unsigned short* Whhbf = Wihbf + 1024 * 32;          // 1024*256

  auto cvt = [&](const float* src, unsigned short* dst, int n) {
    cvt_f32_bf16_kernel<<<(n + 255) / 256, 256, 0, stream>>>(src, dst, n);
  };
  cvt(W_w, Wbf,   256 * 512);
  cvt(U_w, Ubf,   256 * 32);
  cvt(V_w, Vbf,   32 * 256);
  cvt(Wih, Wihbf, 1024 * 32);
  cvt(Whh, Whhbf, 1024 * 256);

  lstm_attn_persistent_kernel<<<BATCH / ROWS, 32, 0, stream>>>(
      x, W_b, U_b, V_b, bih, bhh, fc_w, fc_b,
      Wbf, Ubf, Vbf, Wihbf, Whhbf, out);
}
